// GNN_73667279061015
// MI455X (gfx1250) — matrix operand order repaired and checked by reading, not compiled, not run on hardware
//
#include <hip/hip_runtime.h>

#define N_NODES 50000
#define N_EDGES 400000
#define H 128
#define DEPTH 15
#define NGRAPH 256
#define NF 48
#define EF 13
#define KEI 61
#define BN_EPS 1e-5f

typedef __attribute__((ext_vector_type(16))) __bf16 v16bf;
typedef __attribute__((ext_vector_type(2)))  __bf16 v2bf;
typedef __attribute__((ext_vector_type(8)))  float v8f;
typedef __attribute__((__vector_size__(4 * sizeof(int)))) int v4i;

#define GLOBAL_AS __attribute__((address_space(1)))
#define LDS_AS    __attribute__((address_space(3)))

union Frag { v16bf v; unsigned int u[8]; uint4 q[2]; };

#if __has_builtin(__builtin_amdgcn_global_load_async_to_lds_b128) && \
    __has_builtin(__builtin_amdgcn_s_wait_asynccnt)
#define HAVE_ASYNC_LDS 1
#endif

__device__ __forceinline__ unsigned short f2bf(float f) {
  return __builtin_bit_cast(unsigned short, (__bf16)f);
}
__device__ __forceinline__ unsigned int pk2(float a, float b) {
  v2bf t;
  t[0] = (__bf16)a;
  t[1] = (__bf16)b;
  return __builtin_bit_cast(unsigned int, t);
}
__device__ __forceinline__ float bflo(unsigned int u) {
  return __uint_as_float(u << 16);
}
__device__ __forceinline__ float bfhi(unsigned int u) {
  return __uint_as_float(u & 0xffff0000u);
}

// ---------------------------------------------------------------------------
// Weight prep: swizzle conv / edge-init weights into WMMA *A*-fragment order
// (ISA 16-bit A layout: lane group selects K half-space, VGPR pairs hold
// K = {0,1},{2,3},{4,5},{6,7} then {16..23}, +8 for lanes 16-31), stored as
// frag_base + lane*16 + slot so staging/loads are fully coalesced.
// Also folds Linear bias + BatchNorm into per-channel scale/shift.
// ---------------------------------------------------------------------------
__global__ void __launch_bounds__(256) k_prep(
    const float* __restrict__ convW, const float* __restrict__ convB,
    const float* __restrict__ gamma, const float* __restrict__ beta,
    const float* __restrict__ rmean, const float* __restrict__ rvar,
    const float* __restrict__ Wei,
    unsigned short* __restrict__ wfrag, unsigned short* __restrict__ eifrag,
    float* __restrict__ sc) {
  int i = blockIdx.x * blockDim.x + threadIdx.x;
  const int C0 = DEPTH * 8 * 4 * 512;   // conv A-frag halves
  const int C1 = 8 * 2 * 512;           // edge-init A-frag halves
  const int C2 = DEPTH * H;             // BN scale/shift channels
  if (i < C0) {
    int slot = i & 15, lane = (i >> 4) & 31, kf = (i >> 9) & 3;
    int ht = (i >> 11) & 7, d = i >> 14;
    int row = ht * 16 + (lane & 15);            // output channel h
    int p = slot >> 1;
    int k = kf * 32 + ((p < 4) ? 2 * p : 16 + 2 * (p - 4))
            + ((lane & 16) ? 8 : 0) + (slot & 1);
    wfrag[i] = f2bf(convW[((size_t)d * H + row) * H + k]);
  } else if (i < C0 + C1) {
    int j = i - C0;
    int slot = j & 15, lane = (j >> 4) & 31, kf = (j >> 9) & 1, ht = (j >> 10) & 7;
    int row = ht * 16 + (lane & 15);
    int p = slot >> 1;
    int k = kf * 32 + ((p < 4) ? 2 * p : 16 + 2 * (p - 4))
            + ((lane & 16) ? 8 : 0) + (slot & 1);
    eifrag[j] = (k < KEI) ? f2bf(Wei[row * KEI + k]) : (unsigned short)0;
  } else if (i < C0 + C1 + C2) {
    int j = i - C0 - C1;                // j = d*H + h
    float s = gamma[j] * rsqrtf(rvar[j] + BN_EPS);
    sc[j] = s;
    sc[DEPTH * H + j] = (convB[j] - rmean[j]) * s + beta[j];
  }
}

// ---------------------------------------------------------------------------
// Edge init: ea[e] = relu(concat(x[row[e]], edge_attr[e]) @ Wei^T + bei)
// Computed as z^T = Wei x in^T : A = weight frags (staged in LDS),
// B = input tile from LDS. 128 edges / block, K = 61 padded to 64.
// ---------------------------------------------------------------------------
#define STRIDE_EI 72
__global__ void __launch_bounds__(256) k_edge_init(
    const float* __restrict__ x, const float* __restrict__ eattr,
    const int* __restrict__ eidx, const unsigned short* __restrict__ eifrag,
    const float* __restrict__ bei, unsigned short* __restrict__ ea) {
  __shared__ unsigned short sm[128 * STRIDE_EI];
  __shared__ unsigned short smw[8 * 2 * 512];          // 16KB weight frags
  int base = blockIdx.x * 128;
  int t = threadIdx.x;
  // stage weights to LDS (async if available), 16KB = 256 thr * 4 * 16B
#ifdef HAVE_ASYNC_LDS
  {
    GLOBAL_AS v4i* gsrc = (GLOBAL_AS v4i*)eifrag;
    LDS_AS v4i* ldst = (LDS_AS v4i*)smw;
    #pragma unroll
    for (int i = 0; i < 4; ++i) {
      int q = t + i * 256;
      __builtin_amdgcn_global_load_async_to_lds_b128(gsrc + q, ldst + q, 0, 0);
    }
  }
#else
  {
    const uint4* gw = (const uint4*)eifrag;
    uint4* lw = (uint4*)smw;
    #pragma unroll
    for (int i = 0; i < 4; ++i) lw[t + i * 256] = gw[t + i * 256];
  }
#endif
  {
    int el = t >> 1;
    int k0 = (t & 1) * 32;
    int e = base + el;
    int r = eidx[e];                     // row[e]
    const float* xr = x + (size_t)r * NF;
    const float* er = eattr + (size_t)e * EF;
    #pragma unroll
    for (int kk = 0; kk < 32; ++kk) {
      int k = k0 + kk;
      float v = 0.f;
      if (k < NF) v = xr[k];
      else if (k < KEI) v = er[k - NF];
      sm[el * STRIDE_EI + k] = f2bf(v);
    }
  }
#ifdef HAVE_ASYNC_LDS
  __builtin_amdgcn_s_wait_asynccnt(0);
#endif
  __syncthreads();
  int w = t >> 5, lane = t & 31;
  int erow = (w << 4) + (lane & 15);
  int g16 = (lane & 16) ? 8 : 0;         // B-frag K offset (+8 for lanes 16-31)
  Frag m[2];
  #pragma unroll
  for (int kf = 0; kf < 2; ++kf) {
    const uint4* mp = (const uint4*)&sm[erow * STRIDE_EI + kf * 32 + g16];
    m[kf].q[0] = mp[0];
    m[kf].q[1] = mp[2];
  }
  int e = base + erow;
  int colg = (lane & 16) ? 8 : 0;
  #pragma unroll
  for (int ht = 0; ht < 8; ++ht) {
    v8f acc = {};
    #pragma unroll
    for (int kf = 0; kf < 2; ++kf) {
      Frag aw;
      const uint4* ap = (const uint4*)&smw[(((ht << 1) + kf) << 9) + lane * 16];
      aw.q[0] = ap[0];
      aw.q[1] = ap[1];
      acc = __builtin_amdgcn_wmma_f32_16x16x32_bf16(
          false, aw.v, false, m[kf].v, (short)0, acc, false, false);
    }
    int hc = ht * 16 + colg;             // 8 consecutive h channels from here
    float4 b0 = *(const float4*)(bei + hc);
    float4 b1 = *(const float4*)(bei + hc + 4);
    float z0 = acc[0] + b0.x, z1 = acc[1] + b0.y, z2 = acc[2] + b0.z, z3 = acc[3] + b0.w;
    float z4 = acc[4] + b1.x, z5 = acc[5] + b1.y, z6 = acc[6] + b1.z, z7 = acc[7] + b1.w;
    z0 = z0 > 0.f ? z0 : 0.f; z1 = z1 > 0.f ? z1 : 0.f;
    z2 = z2 > 0.f ? z2 : 0.f; z3 = z3 > 0.f ? z3 : 0.f;
    z4 = z4 > 0.f ? z4 : 0.f; z5 = z5 > 0.f ? z5 : 0.f;
    z6 = z6 > 0.f ? z6 : 0.f; z7 = z7 > 0.f ? z7 : 0.f;
    uint4 st;
    st.x = pk2(z0, z1); st.y = pk2(z2, z3); st.z = pk2(z4, z5); st.w = pk2(z6, z7);
    *(uint4*)(ea + (size_t)e * H + hc) = st;
  }
}

// ---------------------------------------------------------------------------
// Segment sum: amsg[col[e]] += ea[e]   (f32 atomics, 4 channels / thread)
// ---------------------------------------------------------------------------
__global__ void __launch_bounds__(256) k_amsg(
    const unsigned short* __restrict__ ea, const int* __restrict__ eidx,
    float* __restrict__ amsg) {
  long long tid = (long long)blockIdx.x * blockDim.x + threadIdx.x;
  int e = (int)(tid >> 5);
  if (e >= N_EDGES) return;
  int hq = ((int)tid & 31) * 4;
  uint2 d = *(const uint2*)(ea + (size_t)e * H + hq);
  int c = eidx[N_EDGES + e];             // col[e]
  float* dst = amsg + (size_t)c * H + hq;
  unsafeAtomicAdd(dst + 0, bflo(d.x));
  unsafeAtomicAdd(dst + 1, bfhi(d.x));
  unsafeAtomicAdd(dst + 2, bflo(d.y));
  unsafeAtomicAdd(dst + 3, bfhi(d.y));
}

// ---------------------------------------------------------------------------
// Fused layer: m = a_msg[row] - ea[e^1]; z = m @ W^T; ea = 2*relu(z*s + c)
// Computed as z^T = W m^T with weights staged in LDS. 128-edge tiles
// (pairs e^1 stay in-tile) -> in-place update of ea is safe: all global
// reads of the tile complete before the barrier.
// ---------------------------------------------------------------------------
#define STRIDE_L 136
__global__ void __launch_bounds__(256) k_layer(
    unsigned short* __restrict__ ea, const float* __restrict__ amsg,
    const int* __restrict__ eidx, const unsigned short* __restrict__ wfrag_d,
    const float* __restrict__ s_d, const float* __restrict__ c_d) {
  __shared__ unsigned short sm[128 * STRIDE_L];        // 34.0KB m-tile
  __shared__ unsigned short smw[8 * 4 * 512];          // 32KB weight frags
  int base = blockIdx.x * 128;
  int t = threadIdx.x;
  // stage this layer's weight fragments to LDS (32KB = 256 thr * 8 * 16B)
#ifdef HAVE_ASYNC_LDS
  {
    GLOBAL_AS v4i* gsrc = (GLOBAL_AS v4i*)wfrag_d;
    LDS_AS v4i* ldst = (LDS_AS v4i*)smw;
    #pragma unroll
    for (int i = 0; i < 8; ++i) {
      int q = t + i * 256;
      __builtin_amdgcn_global_load_async_to_lds_b128(gsrc + q, ldst + q, 0, 0);
    }
  }
#else
  {
    const uint4* gw = (const uint4*)wfrag_d;
    uint4* lw = (uint4*)smw;
    #pragma unroll
    for (int i = 0; i < 8; ++i) lw[t + i * 256] = gw[t + i * 256];
  }
#endif
  {
    int el = t >> 1;
    int k0 = (t & 1) * 64;
    int e = base + el;
    int r = eidx[e];
    const float4* am = (const float4*)(amsg + (size_t)r * H);
    const unsigned short* rv = ea + (size_t)(e ^ 1) * H;   // reverse edge
    #pragma unroll
    for (int i = 0; i < 8; ++i) {
      int k = k0 + i * 8;
      float4 a0 = am[k >> 2];
      float4 a1 = am[(k >> 2) + 1];
      uint4 r4 = *(const uint4*)(rv + k);
      uint4 pk;
      pk.x = pk2(a0.x - bflo(r4.x), a0.y - bfhi(r4.x));
      pk.y = pk2(a0.z - bflo(r4.y), a0.w - bfhi(r4.y));
      pk.z = pk2(a1.x - bflo(r4.z), a1.y - bfhi(r4.z));
      pk.w = pk2(a1.z - bflo(r4.w), a1.w - bfhi(r4.w));
      *(uint4*)&sm[el * STRIDE_L + k] = pk;
    }
  }
#ifdef HAVE_ASYNC_LDS
  __builtin_amdgcn_s_wait_asynccnt(0);
#endif
  __syncthreads();
  int w = t >> 5, lane = t & 31;
  int erow = (w << 4) + (lane & 15);
  int g16 = (lane & 16) ? 8 : 0;
  Frag m[4];
  #pragma unroll
  for (int kf = 0; kf < 4; ++kf) {
    const uint4* mp = (const uint4*)&sm[erow * STRIDE_L + kf * 32 + g16];
    m[kf].q[0] = mp[0];
    m[kf].q[1] = mp[2];
  }
  int e = base + erow;
  int colg = (lane & 16) ? 8 : 0;
  #pragma unroll
  for (int ht = 0; ht < 8; ++ht) {
    v8f acc = {};
    #pragma unroll
    for (int kf = 0; kf < 4; ++kf) {
      Frag aw;
      const uint4* ap = (const uint4*)&smw[(((ht << 2) + kf) << 9) + lane * 16];
      aw.q[0] = ap[0];
      aw.q[1] = ap[1];
      acc = __builtin_amdgcn_wmma_f32_16x16x32_bf16(
          false, aw.v, false, m[kf].v, (short)0, acc, false, false);
    }
    int hc = ht * 16 + colg;
    float4 s0 = *(const float4*)(s_d + hc);
    float4 s1 = *(const float4*)(s_d + hc + 4);
    float4 c0 = *(const float4*)(c_d + hc);
    float4 c1 = *(const float4*)(c_d + hc + 4);
    float z0 = acc[0] * s0.x + c0.x, z1 = acc[1] * s0.y + c0.y;
    float z2 = acc[2] * s0.z + c0.z, z3 = acc[3] * s0.w + c0.w;
    float z4 = acc[4] * s1.x + c1.x, z5 = acc[5] * s1.y + c1.y;
    float z6 = acc[6] * s1.z + c1.z, z7 = acc[7] * s1.w + c1.w;
    z0 = z0 > 0.f ? z0 + z0 : 0.f; z1 = z1 > 0.f ? z1 + z1 : 0.f;
    z2 = z2 > 0.f ? z2 + z2 : 0.f; z3 = z3 > 0.f ? z3 + z3 : 0.f;
    z4 = z4 > 0.f ? z4 + z4 : 0.f; z5 = z5 > 0.f ? z5 + z5 : 0.f;
    z6 = z6 > 0.f ? z6 + z6 : 0.f; z7 = z7 > 0.f ? z7 + z7 : 0.f;
    uint4 st;
    st.x = pk2(z0, z1); st.y = pk2(z2, z3); st.z = pk2(z4, z5); st.w = pk2(z6, z7);
    *(uint4*)(ea + (size_t)e * H + hc) = st;
  }
}

// ---------------------------------------------------------------------------
// Graph pooling: pooled[batch[n]] += h_node[n]; cnt[batch[n]] += 1
// ---------------------------------------------------------------------------
__global__ void __launch_bounds__(256) k_pool(
    const float* __restrict__ hnode, const int* __restrict__ batch,
    float* __restrict__ pooled, float* __restrict__ cnt) {
  long long tid = (long long)blockIdx.x * blockDim.x + threadIdx.x;
  int n = (int)(tid >> 5);
  if (n >= N_NODES) return;
  int hq = ((int)tid & 31) * 4;
  int g = batch[n];
  float4 v = *(const float4*)(hnode + (size_t)n * H + hq);
  float* dst = pooled + (size_t)g * H + hq;
  unsafeAtomicAdd(dst + 0, v.x);
  unsafeAtomicAdd(dst + 1, v.y);
  unsafeAtomicAdd(dst + 2, v.z);
  unsafeAtomicAdd(dst + 3, v.w);
  if (hq == 0) unsafeAtomicAdd(cnt + g, 1.0f);
}

// ---------------------------------------------------------------------------
// Final FFN + sigmoid: out[g] = sigmoid((pooled[g]/max(cnt,1)) . ffn_W + ffn_b)
// ---------------------------------------------------------------------------
__global__ void __launch_bounds__(128) k_final(
    const float* __restrict__ pooled, const float* __restrict__ cnt,
    const float* __restrict__ fw, const float* __restrict__ fb,
    float* __restrict__ out) {
  __shared__ float red[128];
  int g = blockIdx.x, h = threadIdx.x;
  float c = cnt[g];
  c = c > 1.f ? c : 1.f;
  red[h] = (pooled[(size_t)g * H + h] / c) * fw[h];
  __syncthreads();
  #pragma unroll
  for (int s = 64; s > 0; s >>= 1) {
    if (h < s) red[h] += red[h + s];
    __syncthreads();
  }
  if (h == 0) out[g] = 1.f / (1.f + __expf(-(red[0] + fb[0])));
}

extern "C" void kernel_launch(void* const* d_in, const int* in_sizes, int n_in,
                              void* d_out, int out_size, void* d_ws, size_t ws_size,
                              hipStream_t stream) {
  (void)in_sizes; (void)n_in; (void)out_size; (void)ws_size;
  const float* x     = (const float*)d_in[0];
  const float* eattr = (const float*)d_in[1];
  const int*   eidx  = (const int*)d_in[2];
  const int*   batch = (const int*)d_in[3];
  const float* Wei   = (const float*)d_in[6];
  const float* bei   = (const float*)d_in[7];
  const float* convW = (const float*)d_in[8];
  const float* convB = (const float*)d_in[9];
  const float* gamma = (const float*)d_in[10];
  const float* beta  = (const float*)d_in[11];
  const float* rmean = (const float*)d_in[12];
  const float* rvar  = (const float*)d_in[13];
  const float* fw    = (const float*)d_in[14];
  const float* fb    = (const float*)d_in[15];

  char* ws = (char*)d_ws;
  size_t off = 0;
  auto alloc = [&](size_t bytes) {
    size_t o = off;
    off = (off + bytes + 255) & ~(size_t)255;
    return o;
  };
  unsigned short* ea     = (unsigned short*)(ws + alloc((size_t)N_EDGES * H * 2));
  float*          amsg   = (float*)(ws + alloc((size_t)N_NODES * H * 4));
  float*          pooled = (float*)(ws + alloc((size_t)NGRAPH * H * 4));
  float*          cnt    = (float*)(ws + alloc((size_t)NGRAPH * 4));
  unsigned short* wfrag  = (unsigned short*)(ws + alloc((size_t)DEPTH * 8 * 4 * 512 * 2));
  unsigned short* eifrag = (unsigned short*)(ws + alloc((size_t)8 * 2 * 512 * 2));
  float*          sc     = (float*)(ws + alloc((size_t)2 * DEPTH * H * 4));

  k_prep<<<1000, 256, 0, stream>>>(convW, convB, gamma, beta, rmean, rvar,
                                   Wei, wfrag, eifrag, sc);
  k_edge_init<<<N_EDGES / 128, 256, 0, stream>>>(x, eattr, eidx, eifrag, bei, ea);
  for (int d = 0; d < DEPTH; ++d) {
    (void)hipMemsetAsync(amsg, 0, (size_t)N_NODES * H * 4, stream);
    k_amsg<<<(N_EDGES * 32) / 256, 256, 0, stream>>>(ea, eidx, amsg);
    k_layer<<<N_EDGES / 128, 256, 0, stream>>>(
        ea, amsg, eidx, wfrag + (size_t)d * 8 * 4 * 512,
        sc + (size_t)d * H, sc + (size_t)(DEPTH + d) * H);
  }
  // h_node = segment_sum(h, col)
  (void)hipMemsetAsync(amsg, 0, (size_t)N_NODES * H * 4, stream);
  k_amsg<<<(N_EDGES * 32) / 256, 256, 0, stream>>>(ea, eidx, amsg);
  // global mean pool + FFN
  (void)hipMemsetAsync(pooled, 0, (size_t)NGRAPH * H * 4, stream);
  (void)hipMemsetAsync(cnt, 0, (size_t)NGRAPH * 4, stream);
  k_pool<<<(N_NODES * 32) / 256, 256, 0, stream>>>(amsg, batch, pooled, cnt);
  k_final<<<NGRAPH, 128, 0, stream>>>(pooled, cnt, fw, fb, (float*)d_out);
}
